// MCLinearAttentionBlock_31215822307419
// MI455X (gfx1250) — matrix operand order repaired and checked by reading, not compiled, not run on hardware
//
#include <hip/hip_runtime.h>
#include <hip/hip_bf16.h>
#include <math.h>

// MCLinearAttentionBlock for MI455X (gfx1250, wave32, WMMA).
//
// Roofline: 5 D=1024 GEMMs over 8192 rows = ~86 GFLOP dominate; attention core
// ~11 GFLOP; retrieval ~9 GFLOP. Total HBM traffic (f16 intermediates, f32
// in/out) is a few hundred MB -> memory-bound at 23.3 TB/s (~20-30us), so:
//  * all matmuls via v_wmma_f32_16x16x32_f16 (f32 accum),
//  * intermediates stored f16 (halves traffic; weights/states live in 192MB L2),
//  * h_retrieved done as per-chunk WMMA GEMMs (scalar would be ~9 GFLOP VALU).

typedef _Float16 f16;
typedef __attribute__((ext_vector_type(16))) _Float16 h16;
typedef __attribute__((ext_vector_type(8)))  _Float16 h8;
typedef __attribute__((ext_vector_type(8)))  float    v8f;

#define B_   2
#define T_   4096
#define D_   1024
#define H_   16
#define SEG_ 256
#define N_   16
#define BT_  (B_*T_)
#define EPS_ 1e-6f

// ---- WMMA fragment helpers (layouts per CDNA5 ISA) -------------
// A 16x32 f16: lane half 0 holds K0-7 (lo) + K16-23 (hi); half 1: K8-15 + K24-31.
static __device__ inline h16 mk_frag(h8 lo, h8 hi) {
  h16 r;
#pragma unroll
  for (int i = 0; i < 8; i++) { r[i] = lo[i]; r[i + 8] = hi[i]; }
  return r;
}
static __device__ inline h16 frag_a(const f16* p, int khalf) {
  h8 lo = *(const h8*)(p + khalf * 8);
  h8 hi = *(const h8*)(p + 16 + khalf * 8);
  return mk_frag(lo, hi);
}
// B 32x16 f16: lane = column; lane half 0 holds K0-7 (lo) + K16-23 (hi); half 1: K8-15 + K24-31.
static __device__ inline h16 frag_b(const f16* p, int khalf) {
  h8 lo = *(const h8*)(p + khalf * 8);
  h8 hi = *(const h8*)(p + 16 + khalf * 8);
  return mk_frag(lo, hi);
}
static __device__ inline v8f wmma_f16(h16 a, h16 b, v8f c) {
  return __builtin_amdgcn_wmma_f32_16x16x32_f16(false, a, false, b, (short)0, c,
                                                false, false);
}

// ---- small utility kernels -------------------------------------------------
__global__ void cvt_f32_f16(const float* __restrict__ in, f16* __restrict__ out, int n) {
  int i = blockIdx.x * 256 + threadIdx.x;
  if (i < n) out[i] = (f16)in[i];
}

__global__ __launch_bounds__(256) void rmsnorm_k(const float* __restrict__ x,
                                                 const float* __restrict__ w,
                                                 f16* __restrict__ xn_h,
                                                 f16* __restrict__ x_h) {
  __shared__ float red[256];
  int row = blockIdx.x, tid = threadIdx.x;
  const float* xr = x + (size_t)row * D_;
  float v[4]; float ss = 0.f;
#pragma unroll
  for (int j = 0; j < 4; j++) { v[j] = xr[tid * 4 + j]; ss += v[j] * v[j]; }
  red[tid] = ss; __syncthreads();
  for (int off = 128; off; off >>= 1) { if (tid < off) red[tid] += red[tid + off]; __syncthreads(); }
  float inv = rsqrtf(red[0] * (1.f / D_) + EPS_);
#pragma unroll
  for (int j = 0; j < 4; j++) {
    int d = tid * 4 + j;
    xn_h[(size_t)row * D_ + d] = (f16)(v[j] * inv * w[d]);
    x_h [(size_t)row * D_ + d] = (f16)v[j];
  }
}

// ---- generic NT GEMM: C[M,N] = epilogue(A[M,K] @ W[N,K]^T) -----------------
#define BM 128
#define BN 128
#define KB 32
#define LDT 40  // halves; 80B row stride (multiple of 16B -> aligned b128 LDS reads)

// MODE: 0 = f16 plain, 1 = f16 elu(x)+1, 2 = f32 plain, 3 = f32 + aux (residual)
template <int MODE>
__global__ __launch_bounds__(256) void gemm_nt(const f16* __restrict__ A,
                                               const f16* __restrict__ W,
                                               void* __restrict__ outp,
                                               const float* __restrict__ aux,
                                               int M, int N, int K) {
  __shared__ __attribute__((aligned(16))) f16 sA[BM * LDT];
  __shared__ __attribute__((aligned(16))) f16 sW[BN * LDT];
  const int tid = threadIdx.x;
  const int lane = tid & 31, wave = tid >> 5;
  const int wm = wave >> 1, wn = wave & 1;
  const int lr = lane & 15, khalf = lane >> 4, hio = (lane >> 4) * 8;
  const int mb = M / BM;
  const int m0 = (blockIdx.x % mb) * BM;
  const int n0 = (blockIdx.x / mb) * BN;
  const int ldrow = tid >> 1, ldh = tid & 1;

  v8f acc[2][4] = {};
  for (int k0 = 0; k0 < K; k0 += KB) {
    __syncthreads();
    {
      const uint4* ga = (const uint4*)(A + (size_t)(m0 + ldrow) * K + k0 + ldh * 16);
      uint4* da = (uint4*)(sA + ldrow * LDT + ldh * 16);
      da[0] = ga[0]; da[1] = ga[1];
      const uint4* gw = (const uint4*)(W + (size_t)(n0 + ldrow) * K + k0 + ldh * 16);
      uint4* dw = (uint4*)(sW + ldrow * LDT + ldh * 16);
      dw[0] = gw[0]; dw[1] = gw[1];
    }
    __syncthreads();
    h16 af[2], bf[4];
#pragma unroll
    for (int ms = 0; ms < 2; ms++) af[ms] = frag_a(sA + (wm * 32 + ms * 16 + lr) * LDT, khalf);
#pragma unroll
    for (int ns = 0; ns < 4; ns++) bf[ns] = frag_b(sW + (wn * 64 + ns * 16 + lr) * LDT, khalf);
#pragma unroll
    for (int ms = 0; ms < 2; ms++)
#pragma unroll
      for (int ns = 0; ns < 4; ns++) acc[ms][ns] = wmma_f16(af[ms], bf[ns], acc[ms][ns]);
  }
#pragma unroll
  for (int ms = 0; ms < 2; ms++)
#pragma unroll
    for (int ns = 0; ns < 4; ns++) {
      int col = n0 + wn * 64 + ns * 16 + lr;
#pragma unroll
      for (int r = 0; r < 8; r++) {
        int row = m0 + wm * 32 + ms * 16 + r + hio;
        size_t idx = (size_t)row * N + col;
        float v = acc[ms][ns][r];
        if constexpr (MODE == 0) { ((f16*)outp)[idx] = (f16)v; }
        else if constexpr (MODE == 1) { float e = v > 0.f ? v + 1.f : __expf(v); ((f16*)outp)[idx] = (f16)e; }
        else if constexpr (MODE == 2) { ((float*)outp)[idx] = v; }
        else { ((float*)outp)[idx] = v + aux[idx]; }
      }
    }
}

// ---- chunk states: Sp[(b,n,half,h)][64k][64v] = pk_half^T @ v_half ---------
__global__ __launch_bounds__(256) void schunk_k(const f16* __restrict__ pk,
                                                const f16* __restrict__ vh,
                                                float* __restrict__ Sp) {
  __shared__ __attribute__((aligned(16))) f16 pkT[64 * 136];
  __shared__ __attribute__((aligned(16))) f16 vT [64 * 136];
  int idx = blockIdx.x;
  int h = idx % H_; idx /= H_;
  int half = idx & 1; idx >>= 1;
  int n = idx % N_; int b = idx / N_;
  int tid = threadIdx.x, lane = tid & 31, wave = tid >> 5;
  int lr = lane & 15, khalf = lane >> 4;
  size_t tbase = (size_t)b * T_ + n * SEG_ + half * 128;
  for (int i = 0; i < 32; i++) {  // transpose-stage 128x64 tiles
    int e = i * 256 + tid; int c = e >> 6; int k = e & 63;
    pkT[k * 136 + c] = pk[(tbase + c) * D_ + h * 64 + k];
    vT [k * 136 + c] = vh[(tbase + c) * D_ + h * 64 + k];
  }
  __syncthreads();
  int ms = wave >> 1; int ns0 = (wave & 1) * 2;
  v8f acc[2] = {};
  for (int kk = 0; kk < 4; kk++) {  // K = 128 (this half of the chunk)
    h16 af = frag_a(pkT + (ms * 16 + lr) * 136 + kk * 32, khalf);
#pragma unroll
    for (int j = 0; j < 2; j++) {
      h16 bf = frag_b(vT + ((ns0 + j) * 16 + lr) * 136 + kk * 32, khalf);
      acc[j] = wmma_f16(af, bf, acc[j]);
    }
  }
  float* out = Sp + (size_t)blockIdx.x * 64 * 64;
#pragma unroll
  for (int j = 0; j < 2; j++)
#pragma unroll
    for (int r = 0; r < 8; r++)
      out[(ms * 16 + r + (khalf ? 8 : 0)) * 64 + (ns0 + j) * 16 + lr] = acc[j][r];
}

// ---- prefix over chunks: S_bound (inclusive), S_excl (exclusive) -----------
__global__ void scum_k(const float* __restrict__ Sp, float* __restrict__ Sb,
                       float* __restrict__ Sex) {
  int gid = blockIdx.x * 256 + threadIdx.x;  // B*H*64*64 = 131072
  if (gid >= B_ * H_ * 64 * 64) return;
  int v = gid & 63; int k = (gid >> 6) & 63; int h = (gid >> 12) & (H_ - 1); int b = gid >> 16;
  float run = 0.f;
  for (int n = 0; n < N_; n++) {
    size_t o = ((size_t)((b * N_ + n) * 2 + 0) * H_ + h) * 4096 + k * 64 + v;
    float cur = Sp[o] + Sp[o + (size_t)H_ * 4096];  // half0 + half1
    size_t oo = (((size_t)(b * N_ + n) * H_ + h) * 64 + k) * 64 + v;
    Sex[oo] = run; run += cur; Sb[oo] = run;
  }
}

// ---- z_excl[b,n,h,k] = sum of pk over previous chunks ----------------------
__global__ void zex_k(const f16* __restrict__ pk, float* __restrict__ Zex) {
  int gid = blockIdx.x * 256 + threadIdx.x;  // B*H*64 = 2048
  if (gid >= B_ * H_ * 64) return;
  int k = gid & 63; int h = (gid >> 6) & (H_ - 1); int b = gid >> 10;
  float run = 0.f;
  for (int n = 0; n < N_; n++) {
    Zex[(((size_t)(b * N_ + n) * H_) + h) * 64 + k] = run;
    for (int c = 0; c < SEG_; c++)
      run += (float)pk[((size_t)b * T_ + n * SEG_ + c) * D_ + h * 64 + k];
  }
}

// ---- chunk core: core = (pq@S_excl + tril(pq@pk^T)@v) / denom --------------
// One block per (b,n,half,h); 8 waves, 16 chunk-rows each.
__global__ __launch_bounds__(256) void core_k(const f16* __restrict__ pq,
                                              const f16* __restrict__ pk,
                                              const f16* __restrict__ vh,
                                              const float* __restrict__ Sex,
                                              const float* __restrict__ Zex,
                                              f16* __restrict__ core) {
  __shared__ __attribute__((aligned(16))) f16 vT[64 * 264];    // v transposed
  __shared__ __attribute__((aligned(16))) f16 SexT[64 * 72];   // S_excl^T f16
  __shared__ __attribute__((aligned(16))) f16 Atile[8][16 * 40];
  __shared__ float denomS[128];
  int idx = blockIdx.x;
  int h = idx % H_; idx /= H_;
  int half = idx & 1; idx >>= 1;
  int n = idx % N_; int b = idx / N_;
  int tid = threadIdx.x, lane = tid & 31, wave = tid >> 5;
  int lr = lane & 15, khalf = lane >> 4, hio = khalf * 8;
  size_t tch = (size_t)b * T_ + n * SEG_;
  size_t sbase = ((size_t)(b * N_ + n) * H_ + h) * 4096;
  size_t zbase = ((size_t)(b * N_ + n) * H_ + h) * 64;

  for (int i = 0; i < 64; i++) {  // full chunk of v, transposed
    int e = i * 256 + tid; int c = e >> 6; int v = e & 63;
    vT[v * 264 + c] = vh[(tch + c) * D_ + h * 64 + v];
  }
  for (int i = 0; i < 16; i++) {
    int e = i * 256 + tid; int k = e >> 6; int v = e & 63;
    SexT[v * 72 + k] = (f16)Sex[sbase + k * 64 + v];
  }
  if (tid < 128) denomS[tid] = 0.f;
  __syncthreads();

  int rbase = half * 128 + wave * 16;  // 16 chunk-rows owned by this wave
  h16 aq[2];
#pragma unroll
  for (int ks = 0; ks < 2; ks++)
    aq[ks] = frag_a(pq + (tch + rbase + lr) * D_ + h * 64 + ks * 32, khalf);

  v8f inter[4] = {}; v8f intra[4] = {};
#pragma unroll
  for (int ks = 0; ks < 2; ks++)
#pragma unroll
    for (int vs = 0; vs < 4; vs++) {
      h16 bf = frag_b(SexT + (vs * 16 + lr) * 72 + ks * 32, khalf);
      inter[vs] = wmma_f16(aq[ks], bf, inter[vs]);
    }

  for (int kk = 0; kk < 8; kk++) {  // c' tiles of 32 (uniform: mask handles causality)
    v8f Aacc[2] = {};
#pragma unroll
    for (int ks = 0; ks < 2; ks++)
#pragma unroll
      for (int ns = 0; ns < 2; ns++) {
        h16 bf = frag_b(pk + (tch + kk * 32 + ns * 16 + lr) * D_ + h * 64 + ks * 32, khalf);
        Aacc[ns] = wmma_f16(aq[ks], bf, Aacc[ns]);
      }
    __syncthreads();  // WAR: prior Atile reads done
#pragma unroll
    for (int ns = 0; ns < 2; ns++)
#pragma unroll
      for (int r = 0; r < 8; r++) {
        int rowc = rbase + r + hio;
        int col = kk * 32 + ns * 16 + lr;
        float av = (col <= rowc) ? Aacc[ns][r] : 0.f;
        atomicAdd(&denomS[rowc - half * 128], av);  // row-sum of masked A
        Atile[wave][(r + hio) * 40 + ns * 16 + lr] = (f16)av;
      }
    __syncthreads();
    h16 aA = frag_a(&Atile[wave][lr * 40], khalf);
#pragma unroll
    for (int vs = 0; vs < 4; vs++) {
      h16 bf = frag_b(vT + (vs * 16 + lr) * 264 + kk * 32, khalf);
      intra[vs] = wmma_f16(aA, bf, intra[vs]);
    }
  }
  __syncthreads();
  if (tid < 128) {  // denom += pq . z_excl + eps
    int rowc = half * 128 + tid;
    const f16* pqr = pq + (tch + rowc) * D_ + h * 64;
    float dot = 0.f;
    for (int k = 0; k < 64; k++) dot += (float)pqr[k] * Zex[zbase + k];
    denomS[tid] += dot + EPS_;
  }
  __syncthreads();
#pragma unroll
  for (int vs = 0; vs < 4; vs++)
#pragma unroll
    for (int r = 0; r < 8; r++) {
      int rowc = rbase + r + hio;
      float val = (inter[vs][r] + intra[vs][r]) / denomS[rowc - half * 128];
      core[(tch + rowc) * D_ + h * 64 + vs * 16 + lr] = (f16)val;
    }
}

// ---- chunk means ------------------------------------------------------------
__global__ void mmean_k(const float* __restrict__ x, float* __restrict__ m_stack,
                        float* __restrict__ m_cur) {
  int gid = blockIdx.x * 256 + threadIdx.x;  // B*N*D = 32768
  if (gid >= B_ * N_ * D_) return;
  int d = gid % D_; int n = (gid / D_) % N_; int b = gid / (D_ * N_);
  float acc = 0.f;
  for (int c = 0; c < SEG_; c++) {
    size_t o = ((size_t)b * T_ + n * SEG_ + c) * D_ + d;
    acc += x[o];
    m_cur[o] = acc / (float)(c + 1);
  }
  m_stack[((size_t)b * N_ + n) * D_ + d] = acc * (1.f / SEG_);
}

// ---- retrieval gates (softmax over N cached scores + current) --------------
__global__ __launch_bounds__(128) void gates_k(const float* __restrict__ u,
                                               const float* __restrict__ m_stack,
                                               const float* __restrict__ m_cur,
                                               float* __restrict__ gates,
                                               float* __restrict__ gate_cur) {
  __shared__ float red[128];
  __shared__ float sc[17];
  int bt = blockIdx.x, tid = threadIdx.x;
  int b = bt / T_; int t = bt % T_; int cch = t / SEG_;
  const float* ur = u + (size_t)bt * D_;
  float uu[8];
#pragma unroll
  for (int j = 0; j < 8; j++) uu[j] = ur[tid * 8 + j];
  for (int s = 0; s <= 16; s++) {
    const float* vec = (s < 16) ? (m_stack + ((size_t)b * N_ + s) * D_)
                                : (m_cur + (size_t)bt * D_);
    float p = 0.f;
#pragma unroll
    for (int j = 0; j < 8; j++) p += uu[j] * vec[tid * 8 + j];
    red[tid] = p; __syncthreads();
    for (int off = 64; off; off >>= 1) { if (tid < off) red[tid] += red[tid + off]; __syncthreads(); }
    if (tid == 0) sc[s] = red[0] * (1.f / 32.f);  // 1/sqrt(D)
    __syncthreads();
  }
  if (tid == 0) {
    float mx = sc[16];
    for (int s = 0; s < cch; s++) mx = fmaxf(mx, sc[s]);
    float e[17]; float tot = 0.f;
    for (int s = 0; s < cch; s++) { e[s] = __expf(sc[s] - mx); tot += e[s]; }
    float ec = __expf(sc[16] - mx); tot += ec;
    for (int s = 0; s < N_; s++) gates[(size_t)bt * N_ + s] = (s < cch) ? e[s] / tot : 0.f;
    gate_cur[bt] = ec / tot;
  }
}

// ---- h_retrieved + final combine -------------------------------------------
// One block per (b, chunk, h); hr[t,v] = sum_{n<chunk} g[t,n]*(pq[t]@Sb[n])[v]
__global__ __launch_bounds__(256) void final_k(const f16* __restrict__ pq,
                                               const float* __restrict__ Sb,
                                               const float* __restrict__ gates,
                                               const float* __restrict__ gate_cur,
                                               const float* __restrict__ xout,
                                               float* __restrict__ out) {
  __shared__ __attribute__((aligned(16))) f16 SbT[64 * 72];
  __shared__ float gbuf[256];
  int idx = blockIdx.x;
  int h = idx % H_; idx /= H_;
  int cch = idx % N_; int b = idx / N_;
  int tid = threadIdx.x, lane = tid & 31, wave = tid >> 5;
  int lr = lane & 15, khalf = lane >> 4, hio = khalf * 8;
  size_t tch = (size_t)b * T_ + cch * SEG_;
  int rbase = wave * 32;
  h16 aq[2][2];
#pragma unroll
  for (int ms = 0; ms < 2; ms++)
#pragma unroll
    for (int ks = 0; ks < 2; ks++)
      aq[ms][ks] = frag_a(pq + (tch + rbase + ms * 16 + lr) * D_ + h * 64 + ks * 32, khalf);
  v8f hr[2][4] = {};

  for (int n2 = 0; n2 < cch; n2++) {
    size_t sb = ((size_t)(b * N_ + n2) * H_ + h) * 4096;
    for (int i = 0; i < 16; i++) {
      int e = i * 256 + tid; int k = e >> 6; int v = e & 63;
      SbT[v * 72 + k] = (f16)Sb[sb + k * 64 + v];
    }
    gbuf[tid] = gates[(tch + tid) * N_ + n2];
    __syncthreads();
    v8f Y[2][4] = {};
#pragma unroll
    for (int ks = 0; ks < 2; ks++)
#pragma unroll
      for (int vs = 0; vs < 4; vs++) {
        h16 bf = frag_b(SbT + (vs * 16 + lr) * 72 + ks * 32, khalf);
#pragma unroll
        for (int ms = 0; ms < 2; ms++) Y[ms][vs] = wmma_f16(aq[ms][ks], bf, Y[ms][vs]);
      }
#pragma unroll
    for (int ms = 0; ms < 2; ms++) {
      float g[8];
#pragma unroll
      for (int r = 0; r < 8; r++) g[r] = gbuf[rbase + ms * 16 + r + hio];
#pragma unroll
      for (int vs = 0; vs < 4; vs++)
#pragma unroll
        for (int r = 0; r < 8; r++) hr[ms][vs][r] += g[r] * Y[ms][vs][r];
    }
    __syncthreads();  // WAR before next SbT/gbuf staging
  }
#pragma unroll
  for (int ms = 0; ms < 2; ms++)
#pragma unroll
    for (int r = 0; r < 8; r++) {
      int row = rbase + ms * 16 + r + hio;
      size_t tt = tch + row;
      float gc = gate_cur[tt];
#pragma unroll
      for (int vs = 0; vs < 4; vs++) {
        size_t o = tt * D_ + h * 64 + vs * 16 + lr;
        out[o] = gc * xout[o] + hr[ms][vs][r];
      }
    }
}

// ---- host orchestration -----------------------------------------------------
extern "C" void kernel_launch(void* const* d_in, const int* in_sizes, int n_in,
                              void* d_out, int out_size, void* d_ws, size_t ws_size,
                              hipStream_t stream) {
  (void)in_sizes; (void)n_in; (void)out_size; (void)ws_size;
  const float* x  = (const float*)d_in[0];
  const float* nw = (const float*)d_in[1];
  const float* Wq = (const float*)d_in[2];
  const float* Wk = (const float*)d_in[3];
  const float* Wv = (const float*)d_in[4];
  const float* Wo = (const float*)d_in[5];
  const float* Wu = (const float*)d_in[6];

  char* ws = (char*)d_ws;
  size_t off = 0;
  auto alloc = [&](size_t bytes) -> char* {
    char* p = ws + off; off += (bytes + 255) & ~(size_t)255; return p;
  };
  const size_t WSZ = (size_t)D_ * D_;      // 1M elems per weight
  const size_t XSZ = (size_t)BT_ * D_;     // 8.4M elems
  f16* Wq_h = (f16*)alloc(WSZ * 2);
  f16* Wk_h = (f16*)alloc(WSZ * 2);
  f16* Wv_h = (f16*)alloc(WSZ * 2);
  f16* Wo_h = (f16*)alloc(WSZ * 2);
  f16* Wu_h = (f16*)alloc(WSZ * 2);
  f16* xn_h   = (f16*)alloc(XSZ * 2);
  f16* x_h    = (f16*)alloc(XSZ * 2);
  f16* pq_h   = (f16*)alloc(XSZ * 2);
  f16* pk_h   = (f16*)alloc(XSZ * 2);
  f16* v_h    = (f16*)alloc(XSZ * 2);
  f16* core_h = (f16*)alloc(XSZ * 2);
  float* Sp   = (float*)alloc((size_t)B_ * N_ * 2 * H_ * 4096 * 4);  // per-half partials
  float* Sb   = (float*)alloc((size_t)B_ * N_ * H_ * 4096 * 4);
  float* Sex  = (float*)alloc((size_t)B_ * N_ * H_ * 4096 * 4);
  float* Zex  = (float*)alloc((size_t)B_ * N_ * H_ * 64 * 4);
  float* xout = (float*)alloc(XSZ * 4);
  float* u    = (float*)alloc(XSZ * 4);
  float* mst  = (float*)alloc((size_t)B_ * N_ * D_ * 4);
  float* mcur = (float*)alloc(XSZ * 4);
  float* gts  = (float*)alloc((size_t)BT_ * N_ * 4);
  float* gcur = (float*)alloc((size_t)BT_ * 4);

  const int nW = D_ * D_;
  cvt_f32_f16<<<(nW + 255) / 256, 256, 0, stream>>>(Wq, Wq_h, nW);
  cvt_f32_f16<<<(nW + 255) / 256, 256, 0, stream>>>(Wk, Wk_h, nW);
  cvt_f32_f16<<<(nW + 255) / 256, 256, 0, stream>>>(Wv, Wv_h, nW);
  cvt_f32_f16<<<(nW + 255) / 256, 256, 0, stream>>>(Wo, Wo_h, nW);
  cvt_f32_f16<<<(nW + 255) / 256, 256, 0, stream>>>(Wu, Wu_h, nW);

  rmsnorm_k<<<BT_, 256, 0, stream>>>(x, nw, xn_h, x_h);

  const int gblocks = (BT_ / BM) * (D_ / BN);  // 512
  gemm_nt<1><<<gblocks, 256, 0, stream>>>(xn_h, Wq_h, pq_h, nullptr, BT_, D_, D_);
  gemm_nt<1><<<gblocks, 256, 0, stream>>>(xn_h, Wk_h, pk_h, nullptr, BT_, D_, D_);
  gemm_nt<0><<<gblocks, 256, 0, stream>>>(xn_h, Wv_h, v_h,  nullptr, BT_, D_, D_);

  schunk_k<<<B_ * N_ * 2 * H_, 256, 0, stream>>>(pk_h, v_h, Sp);
  scum_k<<<(B_ * H_ * 4096 + 255) / 256, 256, 0, stream>>>(Sp, Sb, Sex);
  zex_k<<<(B_ * H_ * 64 + 255) / 256, 256, 0, stream>>>(pk_h, Zex);
  core_k<<<B_ * N_ * 2 * H_, 256, 0, stream>>>(pq_h, pk_h, v_h, Sex, Zex, core_h);

  gemm_nt<3><<<gblocks, 256, 0, stream>>>(core_h, Wo_h, xout, x, BT_, D_, D_);
  gemm_nt<2><<<gblocks, 256, 0, stream>>>(x_h, Wu_h, u, nullptr, BT_, D_, D_);

  mmean_k<<<(B_ * N_ * D_ + 255) / 256, 256, 0, stream>>>(x, mst, mcur);
  gates_k<<<BT_, 128, 0, stream>>>(u, mst, mcur, gts, gcur);
  final_k<<<B_ * N_ * H_, 256, 0, stream>>>(pq_h, Sb, gts, gcur, xout, (float*)d_out);
}
